// Encoder_28303834480799
// MI455X (gfx1250) — compile-verified, matrix operand order checked by reading, not run on hardware
//
#include <hip/hip_runtime.h>
#include <hip/hip_bf16.h>

#define B_ 64
#define T_ 128
#define E_ 512
#define H_ 1024
#define NG_ 4096      // 4*H
#define OUTLD_ 2048   // 2*H

typedef __bf16 bf16_t;
typedef __attribute__((ext_vector_type(16))) __bf16 v16bf;
typedef __attribute__((ext_vector_type(8)))  __bf16 v8bf;
typedef __attribute__((ext_vector_type(8)))  float   v8f;

// ---- fragment loader: 16-bit A/B 16x32 WMMA layout --------------------------
// lane L<16 holds row/col (L): K = {k0..k0+7} in elems0-7, {k0+16..k0+23} in 8-15
// lane L>=16 holds row/col (L-16): K = {k0+8..}, {k0+24..}
// caller passes p = rowbase + k0 + (lane>>4)*8  (16B aligned)
__device__ __forceinline__ v16bf ld_frag(const bf16_t* p) {
  union { v16bf v; v8bf h[2]; } u;
  u.h[0] = *(const v8bf*)(p);
  u.h[1] = *(const v8bf*)(p + 16);
  return u.v;
}

__device__ __forceinline__ float fsig(float x)  { return 1.0f / (1.0f + __expf(-x)); }
__device__ __forceinline__ float ftanh(float x) { return 2.0f / (1.0f + __expf(-2.0f * x)) - 1.0f; }

// ---- per-timestep fused dual-direction LSTM cell ---------------------------
// gates = A1 @ W1^T + A2 @ W2^T + bih + bhh ; pointwise LSTM update.
// One wave = one 16x16 (batch x hidden) tile, computing ALL 4 gate tiles so the
// nonlinearity happens in-register with zero cross-lane traffic.
struct StepParams {
  const bf16_t* A1_f; const bf16_t* A1_b; int K1;     // [B, K1] row-major
  const bf16_t* W1_f; const bf16_t* W1_b;             // [4H, K1] row-major
  const bf16_t* A2_f; const bf16_t* A2_b;             // [B, H]  (h_prev)
  const bf16_t* W2_f; const bf16_t* W2_b;             // [4H, H]
  const float*  bih_f; const float* bhh_f;
  const float*  bih_b; const float* bhh_b;
  float*  c_f;   float*  c_b;                          // cell state f32 (in-place)
  float*  h32_f; float*  h32_b;                        // new h, f32 (final-state copy)
  bf16_t* hbf_f; bf16_t* hbf_b;                        // new h, bf16 (next GEMM input)
  float*  out;                                         // null, or out[m*2H + dir*H + n]
};

__global__ __launch_bounds__(256) void lstm_step(StepParams p) {
  const int lane = threadIdx.x & 31;
  const int wave = (blockIdx.x * blockDim.x + threadIdx.x) >> 5; // 0..511
  const int dir  = wave >> 8;        // 0 = forward, 1 = backward
  const int w    = wave & 255;
  const int m0   = (w >> 6) << 4;    // batch tile: 0,16,32,48
  const int n0   = (w & 63) << 4;    // hidden tile: 0..1008
  const int lr   = lane & 15;
  const int half = lane >> 4;
  const int nn   = n0 + lr;

  const bf16_t* A1  = dir ? p.A1_b  : p.A1_f;
  const bf16_t* W1  = dir ? p.W1_b  : p.W1_f;
  const bf16_t* A2  = dir ? p.A2_b  : p.A2_f;
  const bf16_t* W2  = dir ? p.W2_b  : p.W2_f;
  const float*  bih = dir ? p.bih_b : p.bih_f;
  const float*  bhh = dir ? p.bhh_b : p.bhh_f;
  float*  cst = dir ? p.c_b   : p.c_f;
  float*  h32 = dir ? p.h32_b : p.h32_f;
  bf16_t* hbf = dir ? p.hbf_b : p.hbf_f;

  const int K1 = p.K1;

  v8f acc[4];
#pragma unroll
  for (int g = 0; g < 4; ++g) {
    float bv = bih[g * H_ + nn] + bhh[g * H_ + nn];
#pragma unroll
    for (int r = 0; r < 8; ++r) acc[g][r] = bv;
  }

  // ---- input contribution: A1 [B,K1] x W1 [4H,K1]^T ----
  {
    const bf16_t* arow = A1 + (size_t)(m0 + lr) * K1 + half * 8;
    const bf16_t* wrow[4];
#pragma unroll
    for (int g = 0; g < 4; ++g) wrow[g] = W1 + (size_t)(g * H_ + nn) * K1 + half * 8;
    for (int k = 0; k < K1; k += 32) {
      v16bf a = ld_frag(arow + k);
#pragma unroll
      for (int g = 0; g < 4; ++g) {
        v16bf b = ld_frag(wrow[g] + k);
        acc[g] = __builtin_amdgcn_wmma_f32_16x16x32_bf16(
            false, a, false, b, (short)0, acc[g], false, false);
      }
    }
  }
  // ---- recurrent contribution: A2 [B,H] x W2 [4H,H]^T ----
  {
    const bf16_t* arow = A2 + (size_t)(m0 + lr) * H_ + half * 8;
    const bf16_t* wrow[4];
#pragma unroll
    for (int g = 0; g < 4; ++g) wrow[g] = W2 + (size_t)(g * H_ + nn) * H_ + half * 8;
    for (int k = 0; k < H_; k += 32) {
      v16bf a = ld_frag(arow + k);
#pragma unroll
      for (int g = 0; g < 4; ++g) {
        v16bf b = ld_frag(wrow[g] + k);
        acc[g] = __builtin_amdgcn_wmma_f32_16x16x32_bf16(
            false, a, false, b, (short)0, acc[g], false, false);
      }
    }
  }

  // ---- LSTM pointwise: each lane owns 8 (m,n) cells, all 4 gates in-register
  const int mbase = m0 + half * 8;
#pragma unroll
  for (int r = 0; r < 8; ++r) {
    float iv = fsig(acc[0][r]);
    float fv = fsig(acc[1][r]);
    float gv = ftanh(acc[2][r]);
    float ov = fsig(acc[3][r]);
    int idx = (mbase + r) * H_ + nn;
    float cold = cst[idx];
    float cnew = fv * cold + iv * gv;
    float hnew = ov * ftanh(cnew);
    cst[idx] = cnew;
    h32[idx] = hnew;
    hbf[idx] = (bf16_t)hnew;
    if (p.out) p.out[(size_t)(mbase + r) * OUTLD_ + dir * H_ + nn] = hnew;
  }
}

// ---- helpers ---------------------------------------------------------------
__global__ void cvt_f32_bf16(const float* __restrict__ s, bf16_t* __restrict__ d, int n) {
  int i = blockIdx.x * blockDim.x + threadIdx.x;
  if (i < n) d[i] = (bf16_t)s[i];
}

__global__ void embed_gather(const int* __restrict__ idx, const float* __restrict__ tab,
                             bf16_t* __restrict__ ef, bf16_t* __restrict__ eb) {
  int i = blockIdx.x * blockDim.x + threadIdx.x; // over T*B*E
  if (i >= T_ * B_ * E_) return;
  int e = i % E_;
  int tb = i / E_;
  int b = tb % B_;
  int t = tb / B_;
  float v = tab[(size_t)idx[b * T_ + t] * E_ + e];
  ef[((size_t)t * B_ + b) * E_ + e] = (bf16_t)v;
  eb[((size_t)(T_ - 1 - t) * B_ + b) * E_ + e] = (bf16_t)v;
}

__global__ void zero_u32(unsigned* __restrict__ p, long n) {
  long i = (long)blockIdx.x * blockDim.x + threadIdx.x;
  long stride = (long)gridDim.x * blockDim.x;
  for (; i < n; i += stride) p[i] = 0u;
}

__global__ void finalize_states(const float* h0f, const float* h0b,
                                const float* h1f, const float* h1b,
                                const float* c0f, const float* c0b,
                                const float* c1f, const float* c1b,
                                float* __restrict__ out_h, float* __restrict__ out_c) {
  int i = blockIdx.x * blockDim.x + threadIdx.x; // over 2*B*2H
  if (i >= 2 * B_ * OUTLD_) return;
  int n2 = i % OUTLD_;
  int b  = (i / OUTLD_) % B_;
  int l  = i / (OUTLD_ * B_);
  int dir = n2 / H_;
  int n   = n2 % H_;
  const float* hs = l ? (dir ? h1b : h1f) : (dir ? h0b : h0f);
  const float* cs = l ? (dir ? c1b : c1f) : (dir ? c0b : c0f);
  out_h[i] = hs[b * H_ + n];
  out_c[i] = cs[b * H_ + n];
}

extern "C" void kernel_launch(void* const* d_in, const int* in_sizes, int n_in,
                              void* d_out, int out_size, void* d_ws, size_t ws_size,
                              hipStream_t stream) {
  (void)in_sizes; (void)n_in; (void)out_size; (void)ws_size;
  const int*   idx = (const int*)d_in[0];
  const float* tab = (const float*)d_in[1];
  // d_in order: 2..9 = fwd (Wih0,Whh0,bih0,bhh0,Wih1,Whh1,bih1,bhh1); 10..17 = bwd
  const float *srcWih0[2], *srcWhh0[2], *srcWih1[2], *srcWhh1[2];
  const float *bih0[2], *bhh0[2], *bih1[2], *bhh1[2];
  for (int d = 0; d < 2; ++d) {
    int base = 2 + d * 8;
    srcWih0[d] = (const float*)d_in[base + 0];
    srcWhh0[d] = (const float*)d_in[base + 1];
    bih0[d]    = (const float*)d_in[base + 2];
    bhh0[d]    = (const float*)d_in[base + 3];
    srcWih1[d] = (const float*)d_in[base + 4];
    srcWhh1[d] = (const float*)d_in[base + 5];
    bih1[d]    = (const float*)d_in[base + 6];
    bhh1[d]    = (const float*)d_in[base + 7];
  }

  // ---- workspace layout (256B aligned sub-buffers) ----
  char* wp = (char*)d_ws;
  size_t off = 0;
  auto alloc = [&](size_t bytes) -> void* {
    void* p = wp + off;
    off = (off + bytes + 255) & ~(size_t)255;
    return p;
  };
  bf16_t *wih0[2], *whh0[2], *wih1[2], *whh1[2], *emb[2];
  for (int d = 0; d < 2; ++d) {
    wih0[d] = (bf16_t*)alloc((size_t)NG_ * E_ * 2);
    whh0[d] = (bf16_t*)alloc((size_t)NG_ * H_ * 2);
    wih1[d] = (bf16_t*)alloc((size_t)NG_ * H_ * 2);
    whh1[d] = (bf16_t*)alloc((size_t)NG_ * H_ * 2);
  }
  emb[0] = (bf16_t*)alloc((size_t)T_ * B_ * E_ * 2);
  emb[1] = (bf16_t*)alloc((size_t)T_ * B_ * E_ * 2);

  size_t hc_begin = off;
  bf16_t* hbf0[2][2]; bf16_t* hbf1[2][2];   // [dir][pingpong]
  float *h32_0[2], *h32_1[2], *c0[2], *c1[2];
  for (int d = 0; d < 2; ++d)
    for (int pp = 0; pp < 2; ++pp) {
      hbf0[d][pp] = (bf16_t*)alloc((size_t)B_ * H_ * 2);
      hbf1[d][pp] = (bf16_t*)alloc((size_t)B_ * H_ * 2);
    }
  for (int d = 0; d < 2; ++d) {
    h32_0[d] = (float*)alloc((size_t)B_ * H_ * 4);
    h32_1[d] = (float*)alloc((size_t)B_ * H_ * 4);
    c0[d]    = (float*)alloc((size_t)B_ * H_ * 4);
    c1[d]    = (float*)alloc((size_t)B_ * H_ * 4);
  }
  size_t hc_bytes = off - hc_begin;

  // ---- one-time prep: weight f32->bf16 (weights then live in 192MB L2),
  //      embedding gather (fwd + time-reversed), zero h/c states ----
  for (int d = 0; d < 2; ++d) {
    cvt_f32_bf16<<<(NG_ * E_ + 255) / 256, 256, 0, stream>>>(srcWih0[d], wih0[d], NG_ * E_);
    cvt_f32_bf16<<<(NG_ * H_ + 255) / 256, 256, 0, stream>>>(srcWhh0[d], whh0[d], NG_ * H_);
    cvt_f32_bf16<<<(NG_ * H_ + 255) / 256, 256, 0, stream>>>(srcWih1[d], wih1[d], NG_ * H_);
    cvt_f32_bf16<<<(NG_ * H_ + 255) / 256, 256, 0, stream>>>(srcWhh1[d], whh1[d], NG_ * H_);
  }
  embed_gather<<<(T_ * B_ * E_ + 255) / 256, 256, 0, stream>>>(idx, tab, emb[0], emb[1]);
  zero_u32<<<512, 256, 0, stream>>>((unsigned*)(wp + hc_begin), (long)(hc_bytes / 4));

  float* out = (float*)d_out;

  // ---- recurrence: 2 kernels/step (layer0 then layer1), both directions
  //      packed into each launch (512 waves = 64 blocks x 8 waves) ----
  for (int t = 0; t < T_; ++t) {
    int cur = t & 1, prv = cur ^ 1;

    StepParams p0{};
    p0.A1_f = emb[0] + (size_t)t * B_ * E_;
    p0.A1_b = emb[1] + (size_t)t * B_ * E_;
    p0.K1 = E_;
    p0.W1_f = wih0[0]; p0.W1_b = wih0[1];
    p0.A2_f = hbf0[0][prv]; p0.A2_b = hbf0[1][prv];
    p0.W2_f = whh0[0]; p0.W2_b = whh0[1];
    p0.bih_f = bih0[0]; p0.bhh_f = bhh0[0];
    p0.bih_b = bih0[1]; p0.bhh_b = bhh0[1];
    p0.c_f = c0[0]; p0.c_b = c0[1];
    p0.h32_f = h32_0[0]; p0.h32_b = h32_0[1];
    p0.hbf_f = hbf0[0][cur]; p0.hbf_b = hbf0[1][cur];
    p0.out = nullptr;
    lstm_step<<<64, 256, 0, stream>>>(p0);

    StepParams p1{};
    p1.A1_f = hbf0[0][cur]; p1.A1_b = hbf0[1][cur];
    p1.K1 = H_;
    p1.W1_f = wih1[0]; p1.W1_b = wih1[1];
    p1.A2_f = hbf1[0][prv]; p1.A2_b = hbf1[1][prv];
    p1.W2_f = whh1[0]; p1.W2_b = whh1[1];
    p1.bih_f = bih1[0]; p1.bhh_f = bhh1[0];
    p1.bih_b = bih1[1]; p1.bhh_b = bhh1[1];
    p1.c_f = c1[0]; p1.c_b = c1[1];
    p1.h32_f = h32_1[0]; p1.h32_b = h32_1[1];
    p1.hbf_f = hbf1[0][cur]; p1.hbf_b = hbf1[1][cur];
    p1.out = out + (size_t)t * B_ * OUTLD_;   // outputs[t] = concat(hf1, hb1)
    lstm_step<<<64, 256, 0, stream>>>(p1);
  }

  // ---- final (h_t, c_t): (L=2, B, 2H) each, appended after outputs ----
  float* out_h = out + (size_t)T_ * B_ * OUTLD_;
  float* out_c = out_h + (size_t)2 * B_ * OUTLD_;
  finalize_states<<<(2 * B_ * OUTLD_ + 255) / 256, 256, 0, stream>>>(
      h32_0[0], h32_0[1], h32_1[0], h32_1[1], c0[0], c0[1], c1[0], c1[1], out_h, out_c);
}
